// MiMoV2Moe_68753836474420
// MI455X (gfx1250) — compile-verified, matrix operand order checked by reading, not run on hardware
//
#include <hip/hip_runtime.h>
#include <hip/hip_bf16.h>
#include <math.h>

// ---------------------------------------------------------------------------
// MoE (MiMoV2): router(top-4, renorm) + grouped SwiGLU expert GEMMs, bf16 WMMA
// T=4096 D=2048 E=32 F=1024 K=4
// ---------------------------------------------------------------------------

#define T_   4096
#define D_   2048
#define E_   32
#define F_   1024
#define KTOP 4

#define BM 64
#define BN 128
#define BK 32

typedef __attribute__((ext_vector_type(16))) __bf16 v16bf;
typedef __attribute__((ext_vector_type(8)))  __bf16 v8bf;
typedef __attribute__((ext_vector_type(8)))  float  v8f;
typedef __attribute__((ext_vector_type(4)))  int    v4i_t;

// ---- optional gfx1250 async global->LDS path ------------------------------
#if defined(__has_builtin)
#if __has_builtin(__builtin_amdgcn_global_load_async_to_lds_b128)
#define HAVE_ASYNC_LDS 1
#endif
#endif

#if defined(HAVE_ASYNC_LDS)
__device__ inline void async_copy_b128(const void* gsrc, void* ldst) {
    // builtin expects (global int4*, shared int4*, imm offset, imm cpol)
    __builtin_amdgcn_global_load_async_to_lds_b128(
        (__attribute__((address_space(1))) v4i_t*)(unsigned long long)gsrc,
        (__attribute__((address_space(3))) v4i_t*)ldst,
        0, 0);
}
__device__ inline void async_wait0() {
#if __has_builtin(__builtin_amdgcn_s_wait_asynccnt)
    __builtin_amdgcn_s_wait_asynccnt(0);
#else
    asm volatile("s_wait_asynccnt 0x0" ::: "memory");
#endif
}
#endif

__device__ inline v16bf load_frag16(const __bf16* lo, const __bf16* hi) {
    v8bf a = *(const v8bf*)lo;
    v8bf b = *(const v8bf*)hi;
    v16bf r;
#pragma unroll
    for (int i = 0; i < 8; ++i) { r[i] = a[i]; r[8 + i] = b[i]; }
    return r;
}

__device__ inline v8f wmma_bf16(v16bf a, v16bf b, v8f c) {
    return __builtin_amdgcn_wmma_f32_16x16x32_bf16(
        /*neg_a=*/false, a, /*neg_b=*/false, b,
        /*c_mod=*/(short)0, c, /*reuse_a=*/false, /*reuse_b=*/false);
}

// ---------------------------------------------------------------------------
// zero-fill helper (graph-capture safe; out must be re-zeroed every replay)
// ---------------------------------------------------------------------------
__global__ void moe_zero_f(float* __restrict__ p, int n) {
    for (int i = blockIdx.x * blockDim.x + threadIdx.x; i < n;
         i += gridDim.x * blockDim.x)
        p[i] = 0.0f;
}

// ---------------------------------------------------------------------------
// Router: logits = x @ gate_w, softmax, top-4 (renorm), counts histogram.
// ---------------------------------------------------------------------------
__global__ __launch_bounds__(256)
void moe_router(const float* __restrict__ x, const float* __restrict__ gw,
                int* __restrict__ ids_out, float* __restrict__ topw,
                int* __restrict__ counts) {
    __shared__ float xs[D_];
    __shared__ float part[8][E_];
    const int t = blockIdx.x;

    for (int i = threadIdx.x; i < D_; i += 256)
        xs[i] = x[(size_t)t * D_ + i];
    __syncthreads();

    const int e = threadIdx.x & 31;
    const int s = threadIdx.x >> 5;
    {
        float acc = 0.0f;
        const float* col = gw + e;
        const int d0 = s * (D_ / 8), d1 = d0 + (D_ / 8);
        for (int d = d0; d < d1; ++d) acc += xs[d] * col[(size_t)d * E_];
        part[s][e] = acc;
    }
    __syncthreads();

    if (threadIdx.x < 32) {
        float logit = 0.0f;
#pragma unroll
        for (int i = 0; i < 8; ++i) logit += part[i][e];

        float m = logit;
        for (int o = 16; o > 0; o >>= 1) m = fmaxf(m, __shfl_xor(m, o, 32));
        float p = __expf(logit - m);
        float sum = p;
        for (int o = 16; o > 0; o >>= 1) sum += __shfl_xor(sum, o, 32);
        p /= sum;

        float pw = p;
        int   ids[KTOP];
        float wts[KTOP];
        float wsum = 0.0f;
#pragma unroll
        for (int k = 0; k < KTOP; ++k) {
            float v = pw; int idx = e;
            for (int o = 16; o > 0; o >>= 1) {
                float ov = __shfl_xor(v, o, 32);
                int   oi = __shfl_xor(idx, o, 32);
                if (ov > v || (ov == v && oi < idx)) { v = ov; idx = oi; }
            }
            ids[k] = idx; wts[k] = v; wsum += v;
            if (e == idx) pw = -1.0f;
        }
        if (e == 0) {
            const float inv = __builtin_amdgcn_rcpf(wsum);
#pragma unroll
            for (int k = 0; k < KTOP; ++k) {
                ids_out[t * KTOP + k] = ids[k];
                topw[t * KTOP + k]    = wts[k] * inv;
                atomicAdd(&counts[ids[k]], 1);
            }
        }
    }
}

// ---------------------------------------------------------------------------
// Exclusive scan over 32 expert counts (single wave).
// ---------------------------------------------------------------------------
__global__ void moe_scan(const int* __restrict__ counts,
                         int* __restrict__ offsets, int* __restrict__ cursors) {
    const int lane = threadIdx.x;
    const int c = counts[lane];
    int sum = c;
    for (int o = 1; o < 32; o <<= 1) {
        int t = __shfl_up(sum, o, 32);
        if (lane >= o) sum += t;
    }
    offsets[lane] = sum - c;
    cursors[lane] = sum - c;
}

// ---------------------------------------------------------------------------
// Scatter token ids / combine weights into per-expert contiguous lists.
// ---------------------------------------------------------------------------
__global__ __launch_bounds__(256)
void moe_scatter(const int* __restrict__ ids, const float* __restrict__ wts,
                 int* __restrict__ cursors, const int* __restrict__ offsets,
                 int* __restrict__ tok_list, float* __restrict__ wt_list) {
    const int t = blockIdx.x * blockDim.x + threadIdx.x;
    if (t >= T_) return;
#pragma unroll
    for (int k = 0; k < KTOP; ++k) {
        const int   e = ids[t * KTOP + k];
        const float w = wts[t * KTOP + k];
        const int pos = atomicAdd(&cursors[e], 1);
        tok_list[offsets[e] + pos] = t;
        wt_list[offsets[e] + pos]  = w;
    }
}

// ---------------------------------------------------------------------------
// Grouped GEMM #1+#2: h = silu(Xg @ wg) * (Xg @ wu)  (gathered rows, bf16 WMMA)
// Block = 256 thr (8 waves), tile BM=64 x BN=128, K-steps of 32 over D.
// LDS is manually partitioned so the epilogue can reuse it as a 64x128
// staging tile -> coalesced global_store_b128 writeback of h.
// ---------------------------------------------------------------------------
__global__ __launch_bounds__(256)
void moe_gateup(const float* __restrict__ x, const float* __restrict__ wg_all,
                const float* __restrict__ wu_all,
                const int* __restrict__ counts, const int* __restrict__ offsets,
                const int* __restrict__ tok_list, __bf16* __restrict__ hbuf) {
    const int e  = blockIdx.y >> 6;
    const int mt = blockIdx.y & 63;
    const int ne = counts[e];
    if (mt * BM >= ne) return;
    const int off = offsets[e];
    const float* wg = wg_all + (size_t)e * D_ * F_;
    const float* wu = wu_all + (size_t)e * D_ * F_;
    const int nbase0 = blockIdx.x * BN;

    __shared__ __align__(16) unsigned char smem[20480];
    __bf16 (*Xs)[BK]   = (__bf16(*)[BK])(smem);              // [64][32]  4KB
    __bf16 (*Wg_s)[BK] = (__bf16(*)[BK])(smem + 4096);       // [128][32] 8KB
    __bf16 (*Wu_s)[BK] = (__bf16(*)[BK])(smem + 12288);      // [128][32] 8KB
    __bf16 (*Ho)[BN]   = (__bf16(*)[BN])(smem);              // [64][128] 16KB (epilogue)
    __shared__ int toks[BM];

    const int tid = threadIdx.x;
    if (tid < BM) {
        const int r = mt * BM + tid;
        toks[tid] = (r < ne) ? tok_list[off + r] : tok_list[off];  // clamp
    }
    __syncthreads();

    const int wave = tid >> 5, lane = tid & 31;
    const int wm = (wave >> 2) * 32;   // 0 / 32
    const int wn = (wave & 3) * 32;    // 0 / 32 / 64 / 96
    const int half = lane >> 4, l15 = lane & 15;
    const int kb = half * 8;

    v8f accg[2][2] = {{{0}}}, accu[2][2] = {{{0}}};

    for (int kt = 0; kt < D_; kt += BK) {
        // ---- stage X tile (gathered rows, fp32 -> bf16) -------------------
        {
            const int r = tid >> 2;
            const int c = (tid & 3) * 8;
            const float* src = x + (size_t)toks[r] * D_ + kt + c;
            float4 f0 = *(const float4*)(src);
            float4 f1 = *(const float4*)(src + 4);
            v8bf pk;
            pk[0] = (__bf16)f0.x; pk[1] = (__bf16)f0.y;
            pk[2] = (__bf16)f0.z; pk[3] = (__bf16)f0.w;
            pk[4] = (__bf16)f1.x; pk[5] = (__bf16)f1.y;
            pk[6] = (__bf16)f1.z; pk[7] = (__bf16)f1.w;
            *(v8bf*)&Xs[r][c] = pk;
        }
        // ---- stage W tiles, transposed into LDS ---------------------------
        {
            const int kk = tid >> 3;          // 0..31 (k row)
            const int c0 = (tid & 7) * 16;    // 16 F-columns per thread
            const float* g = wg + (size_t)(kt + kk) * F_ + nbase0 + c0;
            const float* u = wu + (size_t)(kt + kk) * F_ + nbase0 + c0;
            __builtin_prefetch(g + (size_t)BK * F_, 0, 1);  // next k-tile
            __builtin_prefetch(u + (size_t)BK * F_, 0, 1);
#pragma unroll
            for (int i = 0; i < 16; i += 4) {
                float4 gv = *(const float4*)(g + i);
                float4 uv = *(const float4*)(u + i);
                Wg_s[c0 + i + 0][kk] = (__bf16)gv.x;
                Wg_s[c0 + i + 1][kk] = (__bf16)gv.y;
                Wg_s[c0 + i + 2][kk] = (__bf16)gv.z;
                Wg_s[c0 + i + 3][kk] = (__bf16)gv.w;
                Wu_s[c0 + i + 0][kk] = (__bf16)uv.x;
                Wu_s[c0 + i + 1][kk] = (__bf16)uv.y;
                Wu_s[c0 + i + 2][kk] = (__bf16)uv.z;
                Wu_s[c0 + i + 3][kk] = (__bf16)uv.w;
            }
        }
        __syncthreads();

        v16bf afrag[2], gfrag[2], ufrag[2];
#pragma unroll
        for (int mi = 0; mi < 2; ++mi) {
            const int r = wm + mi * 16 + l15;
            afrag[mi] = load_frag16(&Xs[r][kb], &Xs[r][16 + kb]);
        }
#pragma unroll
        for (int ni = 0; ni < 2; ++ni) {
            const int c = wn + ni * 16 + l15;
            gfrag[ni] = load_frag16(&Wg_s[c][kb], &Wg_s[c][16 + kb]);
            ufrag[ni] = load_frag16(&Wu_s[c][kb], &Wu_s[c][16 + kb]);
        }
#pragma unroll
        for (int mi = 0; mi < 2; ++mi)
#pragma unroll
            for (int ni = 0; ni < 2; ++ni) {
                accg[mi][ni] = wmma_bf16(afrag[mi], gfrag[ni], accg[mi][ni]);
                accu[mi][ni] = wmma_bf16(afrag[mi], ufrag[ni], accu[mi][ni]);
            }
        __syncthreads();
    }

    // ---- epilogue: h = silu(g)*u -> LDS tile -> coalesced b128 stores -----
#pragma unroll
    for (int mi = 0; mi < 2; ++mi)
#pragma unroll
        for (int ni = 0; ni < 2; ++ni)
#pragma unroll
            for (int i = 0; i < 8; ++i) {
                const int r = wm + mi * 16 + half * 8 + i;
                const int c = wn + ni * 16 + l15;
                const float g = accg[mi][ni][i];
                const float u = accu[mi][ni][i];
                const float h =
                    g * u * __builtin_amdgcn_rcpf(1.0f + __expf(-g));
                Ho[r][c] = (__bf16)h;
            }
    __syncthreads();
    {
        const int r = tid >> 2;            // 64 rows, 4 threads/row
        const int c = (tid & 3) * 32;      // 32 bf16 = 64B per thread
        if (mt * BM + r < ne) {
            v8bf* dst =
                (v8bf*)(hbuf + (size_t)(off + mt * BM + r) * F_ + nbase0 + c);
            const v8bf* srcp = (const v8bf*)&Ho[r][c];
#pragma unroll
            for (int i = 0; i < 4; ++i) dst[i] = srcp[i];
        }
    }
}

// ---------------------------------------------------------------------------
// Grouped GEMM #3: out += c * (h @ wd), scatter-add via f32 global atomics.
// h tile is a pure byte copy -> async global->LDS when available.
// ---------------------------------------------------------------------------
__global__ __launch_bounds__(256)
void moe_down(const __bf16* __restrict__ hbuf, const float* __restrict__ wd_all,
              const int* __restrict__ counts, const int* __restrict__ offsets,
              const int* __restrict__ tok_list, const float* __restrict__ wt_list,
              float* __restrict__ out) {
    const int e  = blockIdx.y >> 6;
    const int mt = blockIdx.y & 63;
    const int ne = counts[e];
    if (mt * BM >= ne) return;
    const int off = offsets[e];
    const float* wd = wd_all + (size_t)e * F_ * D_;
    const int nbase0 = blockIdx.x * BN;  // D columns

    __shared__ __align__(16) __bf16 Hs[BM][BK];    // 4 KB
    __shared__ __align__(16) __bf16 Wd_s[BN][BK];  // 8 KB (transposed)
    __shared__ int   toks[BM];
    __shared__ float cw[BM];

    const int tid = threadIdx.x;
    if (tid < BM) {
        const int r = mt * BM + tid;
        const bool ok = r < ne;
        toks[tid] = ok ? tok_list[off + r] : 0;
        cw[tid]   = ok ? wt_list[off + r] : 0.0f;
    }
    __syncthreads();

    const int wave = tid >> 5, lane = tid & 31;
    const int wm = (wave >> 2) * 32;
    const int wn = (wave & 3) * 32;
    const int half = lane >> 4, l15 = lane & 15;
    const int kb = half * 8;
    const bool full = (mt * BM + BM) <= ne;

    v8f acc[2][2] = {{{0}}};

    for (int kt = 0; kt < F_; kt += BK) {
        // ---- stage h tile (already bf16; contiguous per-expert rows) ------
        {
            const int r = tid >> 2;
            const int c = (tid & 3) * 8;
            const int rg = (mt * BM + r < ne) ? (off + mt * BM + r) : off;
            const __bf16* src = hbuf + (size_t)rg * F_ + kt + c;
#if defined(HAVE_ASYNC_LDS)
            async_copy_b128(src, &Hs[r][c]);
#else
            *(v8bf*)&Hs[r][c] = *(const v8bf*)src;
#endif
        }
        // ---- stage wd tile (fp32 -> bf16, transposed) ---------------------
        {
            const int kk = tid >> 3;
            const int c0 = (tid & 7) * 16;
            const float* w = wd + (size_t)(kt + kk) * D_ + nbase0 + c0;
            __builtin_prefetch(w + (size_t)BK * D_, 0, 1);  // next k-tile
#pragma unroll
            for (int i = 0; i < 16; i += 4) {
                float4 wv = *(const float4*)(w + i);
                Wd_s[c0 + i + 0][kk] = (__bf16)wv.x;
                Wd_s[c0 + i + 1][kk] = (__bf16)wv.y;
                Wd_s[c0 + i + 2][kk] = (__bf16)wv.z;
                Wd_s[c0 + i + 3][kk] = (__bf16)wv.w;
            }
        }
#if defined(HAVE_ASYNC_LDS)
        async_wait0();
#endif
        __syncthreads();

        v16bf afrag[2], bfrag[2];
#pragma unroll
        for (int mi = 0; mi < 2; ++mi) {
            const int r = wm + mi * 16 + l15;
            afrag[mi] = load_frag16(&Hs[r][kb], &Hs[r][16 + kb]);
        }
#pragma unroll
        for (int ni = 0; ni < 2; ++ni) {
            const int c = wn + ni * 16 + l15;
            bfrag[ni] = load_frag16(&Wd_s[c][kb], &Wd_s[c][16 + kb]);
        }
#pragma unroll
        for (int mi = 0; mi < 2; ++mi)
#pragma unroll
            for (int ni = 0; ni < 2; ++ni)
                acc[mi][ni] = wmma_bf16(afrag[mi], bfrag[ni], acc[mi][ni]);
        __syncthreads();
    }

    // ---- epilogue: scale by combine weight, scatter-add into output -------
    if (full) {
#pragma unroll
        for (int mi = 0; mi < 2; ++mi)
#pragma unroll
            for (int ni = 0; ni < 2; ++ni)
#pragma unroll
                for (int i = 0; i < 8; ++i) {
                    const int r = wm + mi * 16 + half * 8 + i;
                    const int col = nbase0 + wn + ni * 16 + l15;
                    atomicAdd(&out[(size_t)toks[r] * D_ + col],
                              acc[mi][ni][i] * cw[r]);
                }
    } else {
#pragma unroll
        for (int mi = 0; mi < 2; ++mi)
#pragma unroll
            for (int ni = 0; ni < 2; ++ni)
#pragma unroll
                for (int i = 0; i < 8; ++i) {
                    const int r = wm + mi * 16 + half * 8 + i;
                    if (mt * BM + r < ne) {
                        const int col = nbase0 + wn + ni * 16 + l15;
                        atomicAdd(&out[(size_t)toks[r] * D_ + col],
                                  acc[mi][ni][i] * cw[r]);
                    }
                }
    }
}

// ---------------------------------------------------------------------------
// Workspace layout (bytes):
//   [0,128)        counts   int[32]
//   [128,256)      offsets  int[32]
//   [256,384)      cursors  int[32]
//   [512, +256KB)  topw     float[T*K]
//   next 256KB     wt_list  float[T*K]
//   next 256KB     tok_list int[T*K]
//   [1MB, +32MB)   hbuf     bf16[T*K * F]
// ---------------------------------------------------------------------------
extern "C" void kernel_launch(void* const* d_in, const int* in_sizes, int n_in,
                              void* d_out, int out_size, void* d_ws, size_t ws_size,
                              hipStream_t stream) {
    (void)in_sizes; (void)n_in; (void)out_size; (void)ws_size;
    const float* x      = (const float*)d_in[0];  // [T, D]
    const float* gate_w = (const float*)d_in[1];  // [D, E]
    const float* w_gate = (const float*)d_in[2];  // [E, D, F]
    const float* w_up   = (const float*)d_in[3];  // [E, D, F]
    const float* w_down = (const float*)d_in[4];  // [E, F, D]

    float* out     = (float*)d_out;                  // [T, D]
    int*   ids_out = (int*)d_out + (size_t)T_ * D_;  // [T, K] (int32 bits)

    char*   ws       = (char*)d_ws;
    int*    counts   = (int*)(ws + 0);
    int*    offsets  = (int*)(ws + 128);
    int*    cursors  = (int*)(ws + 256);
    float*  topw     = (float*)(ws + 512);
    float*  wt_list  = (float*)(ws + 512 + 262144);
    int*    tok_list = (int*)(ws + 512 + 2 * 262144);
    __bf16* hbuf     = (__bf16*)(ws + (1u << 20));

    moe_zero_f<<<1024, 256, 0, stream>>>(out, T_ * D_);
    moe_zero_f<<<1, 128, 0, stream>>>((float*)ws, 128);

    moe_router<<<T_, 256, 0, stream>>>(x, gate_w, ids_out, topw, counts);
    moe_scan<<<1, 32, 0, stream>>>(counts, offsets, cursors);
    moe_scatter<<<T_ / 256, 256, 0, stream>>>(ids_out, topw, cursors, offsets,
                                              tok_list, wt_list);
    moe_gateup<<<dim3(F_ / BN, E_ * 64), 256, 0, stream>>>(
        x, w_gate, w_up, counts, offsets, tok_list, hbuf);
    moe_down<<<dim3(D_ / BN, E_ * 64), 256, 0, stream>>>(
        hbuf, w_down, counts, offsets, tok_list, wt_list, out);
}
